// transformer_566935683726
// MI455X (gfx1250) — matrix operand order repaired and checked by reading, not compiled, not run on hardware
//
#include <hip/hip_runtime.h>

typedef _Float16 h16;
typedef __attribute__((ext_vector_type(16))) _Float16 v16h;
typedef __attribute__((ext_vector_type(8)))  _Float16 v8h;
typedef __attribute__((ext_vector_type(8)))  float    v8f;
typedef __attribute__((ext_vector_type(4)))  unsigned int u32x4;
typedef __attribute__((ext_vector_type(4)))  int i32x4;
typedef __attribute__((ext_vector_type(8)))  int i32x8;

#if __has_builtin(__builtin_amdgcn_tensor_load_to_lds)
#define HAS_TDM 1
#else
#define HAS_TDM 0
#endif

// ---------------------------------------------------------------------------
// WMMA helper: D = A(16x32 f16) * B(32x16 f16) + C(16x16 f32)
// ---------------------------------------------------------------------------
__device__ inline v8f wmma_f16(v16h a, v16h b, v8f c) {
  return __builtin_amdgcn_wmma_f32_16x16x32_f16(
      /*neg_a=*/false, a, /*neg_b=*/false, b,
      /*c_mod=*/(short)0, c, /*reuse_a=*/false, /*reuse_b=*/false);
}

// Load one A-operand (16x32) for this lane from a row-major row pointer.
// Layout: lanes 0-15 (m=lane): halves 0-15 -> k 0-15
//         lanes 16-31        : halves 0-15 -> k 16-31
__device__ inline v16h load_a32(const h16* row, int hi) {
  const int c0 = hi ? 16 : 0;
  v8h lo = *(const v8h*)(row + c0);
  v8h hh = *(const v8h*)(row + c0 + 8);
  v16h r;
#pragma unroll
  for (int i = 0; i < 8; ++i) { r[i] = lo[i]; r[i + 8] = hh[i]; }
  return r;
}

// 16 halves from a 16B-aligned pointer (two v8h reads; safe for LDS rows).
__device__ inline v16h load16(const h16* p) {
  v8h a = *(const v8h*)p;
  v8h b = *(const v8h*)(p + 8);
  v16h r;
#pragma unroll
  for (int i = 0; i < 8; ++i) { r[i] = a[i]; r[i + 8] = b[i]; }
  return r;
}

// ---------------------------------------------------------------------------
// Pack a 64x64 f32 weight into per-lane WMMA B-operand layout (f16).
// ---------------------------------------------------------------------------
__global__ void pack_b(const float* __restrict__ W, h16* __restrict__ dst, int transpose) {
  for (int idx = threadIdx.x + blockIdx.x * blockDim.x; idx < 4096;
       idx += blockDim.x * gridDim.x) {
    int chunk = idx >> 9, rem = idx & 511;
    int lane = rem >> 4, h = rem & 15;
    int kc = chunk >> 2, nb = chunk & 3;
    int n = nb * 16 + (lane & 15);
    int k = kc * 32 + (lane >> 4) * 16 + h;
    float v = transpose ? W[n * 64 + k] : W[k * 64 + n];
    dst[idx] = (h16)v;
  }
}

// ---------------------------------------------------------------------------
// Generic GEMM: out[R,64] = A[R,64](f16) * Wpacked + bias. 8 waves/block,
// one 16x16 WMMA tile per wave, block covers 32 rows x 64 cols.
// ---------------------------------------------------------------------------
template <bool WF32, bool WF16>
__global__ void gemm_rx64(const h16* __restrict__ A, const h16* __restrict__ Bp,
                          const float* __restrict__ bias,
                          float* __restrict__ outF, h16* __restrict__ outH) {
  const int tid = threadIdx.x;
  const int wv = tid >> 5, lane = tid & 31, l15 = lane & 15, hi = lane >> 4;
  const int mt = wv >> 2, nt = wv & 3;
  const int row0 = blockIdx.x * 32 + mt * 16;

  const h16* arow = A + (size_t)(row0 + l15) * 64;
  v16h a0 = load_a32(arow, hi);        // k 0..31
  v16h a1 = load_a32(arow + 32, hi);   // k 32..63
  v16h b0 = *(const v16h*)(Bp + (size_t)(0 + nt) * 512 + lane * 16);
  v16h b1 = *(const v16h*)(Bp + (size_t)(4 + nt) * 512 + lane * 16);

  v8f c = {};
  c = wmma_f16(a0, b0, c);
  c = wmma_f16(a1, b1, c);

  const int col = nt * 16 + l15;
  const float bv = bias[col];
#pragma unroll
  for (int r = 0; r < 8; ++r) {
    int row = row0 + r + hi * 8;              // C layout: lanes16-31 -> M=r+8
    float v = c[r] + bv;
    size_t off = (size_t)row * 64 + col;
    if constexpr (WF32) outF[off] = v;
    if constexpr (WF16) outH[off] = (h16)v;
  }
}

// ---------------------------------------------------------------------------
// GroupNorm stats: one block per (b, group); mean & rstd over 16ch x 4096 sp.
// ---------------------------------------------------------------------------
__global__ void gn_stats(const float* __restrict__ x, float* __restrict__ stats) {
  const int bg = blockIdx.x, b = bg >> 2, g = bg & 3;
  __shared__ float s_sum[256], s_sq[256];
  float sum = 0.f, sq = 0.f;
  for (int t = threadIdx.x; t < 65536; t += 256) {
    int sp = t >> 4, cc = g * 16 + (t & 15);
    float v = x[((size_t)b * 4096 + sp) * 64 + cc];
    sum += v; sq += v * v;
  }
  s_sum[threadIdx.x] = sum; s_sq[threadIdx.x] = sq;
  __syncthreads();
  for (int s = 128; s > 0; s >>= 1) {
    if (threadIdx.x < s) {
      s_sum[threadIdx.x] += s_sum[threadIdx.x + s];
      s_sq[threadIdx.x]  += s_sq[threadIdx.x + s];
    }
    __syncthreads();
  }
  if (threadIdx.x == 0) {
    float mu = s_sum[0] * (1.f / 65536.f);
    float var = s_sq[0] * (1.f / 65536.f) - mu * mu;
    stats[bg * 2]     = mu;
    stats[bg * 2 + 1] = rsqrtf(var + 1e-5f);
  }
}

// ---------------------------------------------------------------------------
// GN-normalize + 1x1 conv: xs[b, o*64+hh, ww] = sum_c xn[b,c,hh,ww]*W1[o,c]+b1[o]
// ---------------------------------------------------------------------------
__global__ void gn_conv1(const float* __restrict__ x, const float* __restrict__ stats,
                         const float* __restrict__ gng, const float* __restrict__ gnb,
                         const float* __restrict__ w1, const float* __restrict__ b1,
                         float* __restrict__ xsF, h16* __restrict__ xsH) {
  const int hh = blockIdx.x, b = blockIdx.y;
  __shared__ float xn[64 * 65];
  for (int t = threadIdx.x; t < 4096; t += 256) {
    int c = t & 63, ww = t >> 6;
    float v = x[(((size_t)b * 64 + hh) * 64 + ww) * 64 + c];
    int g = c >> 4;
    float mu = stats[(b * 4 + g) * 2], rs = stats[(b * 4 + g) * 2 + 1];
    xn[c * 65 + ww] = (v - mu) * rs * gng[c] + gnb[c];
  }
  __syncthreads();
  for (int t = threadIdx.x; t < 4096; t += 256) {
    int o = t >> 6, ww = t & 63;
    float acc = b1[o];
#pragma unroll 8
    for (int c = 0; c < 64; ++c) acc += w1[o * 64 + c] * xn[c * 65 + ww];
    size_t off = ((size_t)b * 4096 + o * 64 + hh) * 64 + ww;
    xsF[off] = acc; xsH[off] = (h16)acc;
  }
}

// V[b,i,h*32+d] -> VT[b,h,d,i]
__global__ void vtrans(const h16* __restrict__ v, h16* __restrict__ vt) {
  int idx = blockIdx.x * 256 + threadIdx.x;       // 1048576
  int i = idx & 4095, d = (idx >> 12) & 31, hd = (idx >> 17) & 1, b = idx >> 18;
  vt[idx] = v[((size_t)(b * 4096 + i)) * 64 + hd * 32 + d];
}

// ---------------------------------------------------------------------------
// Flash self-attention.  Each wave owns 16 query rows (wave-autonomous online
// softmax); the 32-key x 32-dh K tile is shared per block and staged into LDS
// by the Tensor Data Mover, double-buffered (TENSORcnt + workgroup barrier).
// TDM pad fields give the LDS tile an 80B row pitch (bank de-conflict).
// grid = (4096/128, B*HEADS), block = 256 (8 waves).
// ---------------------------------------------------------------------------
__global__ void flash_self(const h16* __restrict__ Q, const h16* __restrict__ K,
                           const h16* __restrict__ VT, h16* __restrict__ Z) {
  const int tid = threadIdx.x, wv = tid >> 5, lane = tid & 31;
  const int l15 = lane & 15, hi = lane >> 4;
  const int bh = blockIdx.y, b = bh >> 1, head = bh & 1;
  const int q0 = blockIdx.x * 128 + wv * 16;

  __shared__ __align__(16) h16 lds_p[8][16][40];   // per-wave P tile (16x32, padded)
  __shared__ __align__(16) h16 kbuf[2][32][40];    // K tiles: 32 keys x 32 dh, 80B pitch

  const h16* qrow = Q + ((size_t)(b * 4096 + q0 + l15)) * 64 + head * 32;
  v16h qa = load_a32(qrow, hi);                    // resident A operand (16x32)

  v8f acc0 = {}, acc1 = {};
  float mrow[8], lrow[8];
#pragma unroll
  for (int r = 0; r < 8; ++r) { mrow[r] = -1e30f; lrow[r] = 0.f; }

  const h16* kglob = K + (size_t)b * 4096 * 64 + head * 32;  // tile column base
  const h16* vbase = VT + ((size_t)(b * 2 + head)) * 32 * 4096;

#if HAS_TDM
  auto tdm_issue = [&](int buf, int kb) {
    unsigned lds = (unsigned)(unsigned long long)(uintptr_t)&kbuf[buf][0][0];
    unsigned long long ga =
        (unsigned long long)(uintptr_t)(kglob + (size_t)kb * 64);
    u32x4 g0;
    g0[0] = 1u;                                   // count=1, user descriptor
    g0[1] = lds;                                  // lds_addr
    g0[2] = (unsigned)ga;                         // global_addr[31:0]
    g0[3] = (unsigned)(ga >> 32) | (2u << 30);    // global_addr[56:32] | type=2
    i32x8 g1;
    g1[0] = 0x00010000 |                          // data_size = 2 bytes
            (1 << 20) | (3 << 22) | (3 << 25);    // pad: every 16 DW, +4 DW
    g1[1] = (64 << 16);                           // tensor_dim0 = 64 elems
    g1[2] = (4096 << 16);                         // tensor_dim1 = 4096 rows
    g1[3] = (32 << 16);                           // tile_dim0 = 32 elems
    g1[4] = 32;                                   // tile_dim1 = 32 keys
    g1[5] = 64;                                   // tensor_dim0_stride = 64
    g1[6] = 0; g1[7] = 0;
    i32x4 z4 = {};
#if defined(__clang_major__) && __clang_major__ >= 23
    i32x8 z8 = {};
    __builtin_amdgcn_tensor_load_to_lds(g0, g1, z4, z4, z8, 0);
#else
    __builtin_amdgcn_tensor_load_to_lds(g0, g1, z4, z4, 0);
#endif
  };
  if (wv == 0) tdm_issue(0, 0);
#endif

  for (int kb = 0; kb < 4096; kb += 32) {
    int cur = (kb >> 5) & 1;
#if HAS_TDM
    if (wv == 0) {
      if (kb + 32 < 4096) {
        tdm_issue(cur ^ 1, kb + 32);              // prefetch next tile
        __builtin_amdgcn_s_wait_tensorcnt(1);     // wait for current tile only
      } else {
        __builtin_amdgcn_s_wait_tensorcnt(0);
      }
    }
    __syncthreads();
#else
    cur = 0;
    __syncthreads();
    for (int t = tid; t < 1024; t += 256) {       // cooperative fallback stage
      int key = t >> 5, d = t & 31;
      kbuf[0][key][d] = kglob[(size_t)(kb + key) * 64 + d];
    }
    __syncthreads();
#endif
    if (kb + 32 < 4096)   // gfx1250 global_prefetch_b8 for next V block
      __builtin_prefetch((const void*)(vbase + (size_t)l15 * 4096 + kb + 32), 0, 1);

    // S = Q * K^T  (B-operand from the shared LDS K tile; contiguous per lane)
    v16h kb0 = load16(&kbuf[cur][l15][hi * 16]);
    v16h kb1 = load16(&kbuf[cur][16 + l15][hi * 16]);
    v8f zc = {};
    v8f s0 = wmma_f16(qa, kb0, zc);
    v8f s1 = wmma_f16(qa, kb1, zc);

#pragma unroll
    for (int r = 0; r < 8; ++r) {
      float a  = s0[r] * 0.125f;                   // / sqrt(hid)=8
      float c2 = s1[r] * 0.125f;
      float mx = fmaxf(a, c2);
#pragma unroll
      for (int d = 1; d < 16; d <<= 1) mx = fmaxf(mx, __shfl_xor(mx, d, 32));
      float mnew = fmaxf(mrow[r], mx);
      float corr = __expf(mrow[r] - mnew);
      float p0 = __expf(a - mnew);
      float p1 = __expf(c2 - mnew);
      float rs = p0 + p1;
#pragma unroll
      for (int d = 1; d < 16; d <<= 1) rs += __shfl_xor(rs, d, 32);
      lrow[r] = lrow[r] * corr + rs;
      mrow[r] = mnew;
      acc0[r] *= corr; acc1[r] *= corr;
      int m = r + hi * 8;                          // C layout row
      lds_p[wv][m][l15]      = (h16)p0;
      lds_p[wv][m][16 + l15] = (h16)p1;
    }
    // Same-wave DS ops are in-order on CDNA5; stop compiler reordering only.
    asm volatile("" ::: "memory");

    // Reload P in A-operand layout, then Z += P * V  (K = 32 keys).
    v16h pa  = load_a32(&lds_p[wv][l15][0], hi);
    v16h va0 = *(const v16h*)(vbase + (size_t)l15 * 4096 + kb + hi * 16);
    v16h va1 = *(const v16h*)(vbase + (size_t)(16 + l15) * 4096 + kb + hi * 16);
    acc0 = wmma_f16(pa, va0, acc0);
    acc1 = wmma_f16(pa, va1, acc1);

    __syncthreads();   // all waves done with kbuf[cur] before TDM refills it
  }

#pragma unroll
  for (int r = 0; r < 8; ++r) {
    int row = q0 + r + hi * 8;
    float inv = 1.f / lrow[r];
    size_t off = ((size_t)(b * 4096 + row)) * 64 + head * 32;
    Z[off + l15]      = (h16)(acc0[r] * inv);
    Z[off + 16 + l15] = (h16)(acc1[r] * inv);
  }
}

// Cross attention with 8 slots: tiny, one thread per token.
__global__ void cross_attn(const float* __restrict__ Qf, const float* __restrict__ Kf,
                           const float* __restrict__ Vf, h16* __restrict__ Z) {
  int idx = blockIdx.x * 256 + threadIdx.x;
  if (idx >= 16384) return;
  int b = idx >> 12;
  const float* q = Qf + (size_t)idx * 64;
  for (int hd = 0; hd < 2; ++hd) {
    float sc[8]; float mx = -1e30f;
#pragma unroll
    for (int s = 0; s < 8; ++s) {
      const float* kk = Kf + (size_t)(b * 8 + s) * 64 + hd * 32;
      float a = 0.f;
      for (int d = 0; d < 32; ++d) a += q[hd * 32 + d] * kk[d];
      sc[s] = a * 0.125f;
      mx = fmaxf(mx, sc[s]);
    }
    float den = 0.f;
#pragma unroll
    for (int s = 0; s < 8; ++s) { sc[s] = __expf(sc[s] - mx); den += sc[s]; }
    float invd = 1.f / den;
    for (int d = 0; d < 32; ++d) {
      float z = 0.f;
#pragma unroll
      for (int s = 0; s < 8; ++s) z += sc[s] * Vf[(size_t)(b * 8 + s) * 64 + hd * 32 + d];
      Z[(size_t)idx * 64 + hd * 32 + d] = (h16)(z * invd);
    }
  }
}

// out = [gelu?](LN(inp)*g + b + res); res may alias outF (read-before-write).
__global__ void ln_res(const float* __restrict__ inp, const float* res,
                       const float* __restrict__ g, const float* __restrict__ bb,
                       float* outF, h16* __restrict__ outH, int gelu) {
  int row = blockIdx.x * 256 + threadIdx.x;
  if (row >= 16384) return;
  const float* p = inp + (size_t)row * 64;
  float buf[64]; float mu = 0.f;
  for (int j = 0; j < 64; ++j) { buf[j] = p[j]; mu += buf[j]; }
  mu *= (1.f / 64.f);
  float var = 0.f;
  for (int j = 0; j < 64; ++j) { float d = buf[j] - mu; var += d * d; }
  float rs = rsqrtf(var * (1.f / 64.f) + 1e-5f);
  const float* rp = res + (size_t)row * 64;
  for (int j = 0; j < 64; ++j)
    buf[j] = (buf[j] - mu) * rs * g[j] + bb[j] + rp[j];
  for (int j = 0; j < 64; ++j) {
    float y = buf[j];
    if (gelu) y = 0.5f * y * (1.f + erff(y * 0.70710678f));
    size_t off = (size_t)row * 64 + j;
    outF[off] = y; outH[off] = (h16)y;
  }
}

__global__ void f32_to_h16(const float* __restrict__ src, h16* __restrict__ dst, int n) {
  int i = blockIdx.x * 256 + threadIdx.x;
  if (i < n) dst[i] = (h16)src[i];
}

// ---------------------------------------------------------------------------
extern "C" void kernel_launch(void* const* d_in, const int* in_sizes, int n_in,
                              void* d_out, int out_size, void* d_ws, size_t ws_size,
                              hipStream_t stream) {
  const float* x       = (const float*)d_in[0];
  const float* slots   = (const float*)d_in[1];
  const float* conv1_w = (const float*)d_in[2];
  const float* conv1_b = (const float*)d_in[3];
  const float* conv2_w = (const float*)d_in[4];
  const float* conv2_b = (const float*)d_in[5];
  const float* gn_g    = (const float*)d_in[6];
  const float* gn_b    = (const float*)d_in[7];
  const float* ln2_g   = (const float*)d_in[8];
  const float* ln2_b   = (const float*)d_in[9];
  const float* ln3_g   = (const float*)d_in[10];
  const float* ln3_b   = (const float*)d_in[11];
  const float* ln4_g   = (const float*)d_in[12];
  const float* ln4_b   = (const float*)d_in[13];
  const float* sa_qw   = (const float*)d_in[14];
  const float* sa_qb   = (const float*)d_in[15];
  const float* sa_kw   = (const float*)d_in[16];
  const float* sa_kb   = (const float*)d_in[17];
  const float* sa_vw   = (const float*)d_in[18];
  const float* sa_vb   = (const float*)d_in[19];
  const float* sa_ow   = (const float*)d_in[20];
  const float* sa_ob   = (const float*)d_in[21];
  const float* ca_qw   = (const float*)d_in[22];
  const float* ca_qb   = (const float*)d_in[23];
  const float* ca_kw   = (const float*)d_in[24];
  const float* ca_kb   = (const float*)d_in[25];
  const float* ca_vw   = (const float*)d_in[26];
  const float* ca_vb   = (const float*)d_in[27];
  const float* ca_ow   = (const float*)d_in[28];
  const float* ca_ob   = (const float*)d_in[29];
  const float* ff1_w   = (const float*)d_in[30];
  const float* ff1_b   = (const float*)d_in[31];

  char* ws = (char*)d_ws;
  const size_t MB = 1u << 20;
  float* xsF   = (float*)(ws + 0 * MB);        // 4 MB f32 token features
  float* tmp   = (float*)(ws + 4 * MB);        // 4 MB f32 GEMM scratch
  h16*   xsH   = (h16*)  (ws + 8 * MB);        // 2 MB f16 token features
  h16*   qh    = (h16*)  (ws + 10 * MB);
  h16*   kh    = (h16*)  (ws + 12 * MB);
  h16*   vh    = (h16*)  (ws + 14 * MB);
  h16*   vt    = (h16*)  (ws + 16 * MB);
  h16*   zh    = (h16*)  (ws + 18 * MB);
  float* cak   = (float*)(ws + 20 * MB);
  float* cav   = cak + 2048;
  h16*   slotsH= (h16*)(cav + 2048);
  float* stats = (float*)((char*)slotsH + 4096);
  h16*   packs = (h16*)(ws + 20 * MB + 65536); // 10 x 8 KB packed weights

  auto P = [&](int i) { return packs + (size_t)i * 4096; };

  pack_b<<<4, 256, 0, stream>>>(sa_qw, P(0), 0);
  pack_b<<<4, 256, 0, stream>>>(sa_kw, P(1), 0);
  pack_b<<<4, 256, 0, stream>>>(sa_vw, P(2), 0);
  pack_b<<<4, 256, 0, stream>>>(sa_ow, P(3), 0);
  pack_b<<<4, 256, 0, stream>>>(ca_qw, P(4), 0);
  pack_b<<<4, 256, 0, stream>>>(ca_kw, P(5), 0);
  pack_b<<<4, 256, 0, stream>>>(ca_vw, P(6), 0);
  pack_b<<<4, 256, 0, stream>>>(ca_ow, P(7), 0);
  pack_b<<<4, 256, 0, stream>>>(ff1_w, P(8), 0);
  pack_b<<<4, 256, 0, stream>>>(conv2_w, P(9), 1);   // conv2: B[k][n]=W[n][k]
  f32_to_h16<<<8, 256, 0, stream>>>(slots, slotsH, 2048);

  gn_stats<<<16, 256, 0, stream>>>(x, stats);
  gn_conv1<<<dim3(64, 4), 256, 0, stream>>>(x, stats, gn_g, gn_b, conv1_w, conv1_b, xsF, xsH);

  // Self-attention: QKV projections (WMMA), TDM-staged flash (WMMA), out proj.
  gemm_rx64<false, true><<<512, 256, 0, stream>>>(xsH, P(0), sa_qb, nullptr, qh);
  gemm_rx64<false, true><<<512, 256, 0, stream>>>(xsH, P(1), sa_kb, nullptr, kh);
  gemm_rx64<false, true><<<512, 256, 0, stream>>>(xsH, P(2), sa_vb, nullptr, vh);
  vtrans<<<4096, 256, 0, stream>>>(vh, vt);
  flash_self<<<dim3(32, 8), 256, 0, stream>>>(qh, kh, vt, zh);
  gemm_rx64<true, false><<<512, 256, 0, stream>>>(zh, P(3), sa_ob, tmp, nullptr);
  ln_res<<<64, 256, 0, stream>>>(tmp, xsF, ln2_g, ln2_b, xsF, xsH, 1);  // gelu

  // Cross-attention with 8 slots.
  gemm_rx64<true, false><<<512, 256, 0, stream>>>(xsH, P(4), ca_qb, tmp, nullptr);
  gemm_rx64<true, false><<<1, 256, 0, stream>>>(slotsH, P(5), ca_kb, cak, nullptr);
  gemm_rx64<true, false><<<1, 256, 0, stream>>>(slotsH, P(6), ca_vb, cav, nullptr);
  cross_attn<<<64, 256, 0, stream>>>(tmp, cak, cav, zh);
  gemm_rx64<true, false><<<512, 256, 0, stream>>>(zh, P(7), ca_ob, tmp, nullptr);
  ln_res<<<64, 256, 0, stream>>>(tmp, xsF, ln3_g, ln3_b, xsF, xsH, 0);

  // FFN + LN + residual.
  gemm_rx64<true, false><<<512, 256, 0, stream>>>(xsH, P(8), ff1_b, tmp, nullptr);
  ln_res<<<64, 256, 0, stream>>>(tmp, xsF, ln4_g, ln4_b, xsF, xsH, 0);

  // conv2 (feature-dim matmul); output layout matches [b,h,w,f] directly.
  gemm_rx64<true, false><<<512, 256, 0, stream>>>(xsH, P(9), conv2_b, (float*)d_out, nullptr);
}
